// PointsFusion2_5291399708687
// MI455X (gfx1250) — matrix operand order repaired and checked by reading, not compiled, not run on hardware
//
#include <hip/hip_runtime.h>
#include <hip/hip_bf16.h>
#include <math.h>

#define BB 8
#define NN 4096
#define N1 2048
#define N2 2048
#define NB 64          // neighbors per point (2k)
#define CO 64          // C_OUT
#define KP 32          // padded K for the projection GEMM (C_IN=4 + bias col + zeros)

typedef __attribute__((ext_vector_type(16))) _Float16 v16h;
typedef __attribute__((ext_vector_type(8)))  float    v8f;

union Frag16 { v16h h; unsigned u[8]; };

// ---------------- prep: transpose + gather fused candidate set ----------------
__global__ void prep_kernel(const float* __restrict__ p1, const float* __restrict__ p2,
                            const float* __restrict__ p3,
                            const int* __restrict__ ridx1, const int* __restrict__ ridx2,
                            float* __restrict__ p1t, float* __restrict__ p2t,
                            float* __restrict__ p3t, float* __restrict__ newp) {
  const int b = blockIdx.y;
  const int n = blockIdx.x * 256 + threadIdx.x;
  const long ib = (long)b * 3 * NN;
  const long ot = ((long)b * NN + n) * 3;
#pragma unroll
  for (int c = 0; c < 3; ++c) {
    p1t[ot + c] = p1[ib + (long)c * NN + n];
    p2t[ot + c] = p2[ib + (long)c * NN + n];
    p3t[ot + c] = p3[ib + (long)c * NN + n];
  }
  if (n < N1) {
    const int i = ridx1[b * N1 + n];
#pragma unroll
    for (int c = 0; c < 3; ++c) newp[ot + c] = p1[ib + (long)c * NN + i];
  } else {
    const int i = ridx2[b * N2 + (n - N1)];
#pragma unroll
    for (int c = 0; c < 3; ++c) newp[ot + c] = p2[ib + (long)c * NN + i];
  }
}

// ---------------- knn: one thread per query, db tiled through LDS ----------------
template <int K>
__global__ __launch_bounds__(128)
void knn_kernel(const float* __restrict__ newp, const float* __restrict__ dbt,
                float* __restrict__ feats, float* __restrict__ grouped, int off) {
  const int b = blockIdx.y;
  const int n = blockIdx.x * 128 + threadIdx.x;
  __shared__ float tile[128 * 3];

  const long qb = ((long)b * NN + n) * 3;
  const float qx = newp[qb + 0], qy = newp[qb + 1], qz = newp[qb + 2];
  const float* __restrict__ db = dbt + (long)b * NN * 3;

  float bestD[K];
  int   bestI[K];
#pragma unroll
  for (int t = 0; t < K; ++t) { bestD[t] = 3.0e38f; bestI[t] = 0; }
  float worst = 3.0e38f;
  int worstPos = 0;

  for (int t0 = 0; t0 < NN; t0 += 128) {
    __syncthreads();
    {
      const int p = t0 + threadIdx.x;
      tile[threadIdx.x * 3 + 0] = db[p * 3 + 0];
      tile[threadIdx.x * 3 + 1] = db[p * 3 + 1];
      tile[threadIdx.x * 3 + 2] = db[p * 3 + 2];
    }
    __syncthreads();
    if (t0 + 128 < NN) __builtin_prefetch(&db[(t0 + 128) * 3], 0, 0); // global_prefetch_b8

    for (int j = 0; j < 128; ++j) {
      const float dx = tile[j * 3 + 0] - qx;
      const float dy = tile[j * 3 + 1] - qy;
      const float dz = tile[j * 3 + 2] - qz;
      const float d2 = dx * dx + dy * dy + dz * dz;
      if (d2 < worst) {
        const int idx = t0 + j;
#pragma unroll
        for (int t = 0; t < K; ++t)
          if (t == worstPos) { bestD[t] = d2; bestI[t] = idx; }
        worst = -1.0f;
#pragma unroll
        for (int t = 0; t < K; ++t)
          if (bestD[t] > worst) { worst = bestD[t]; worstPos = t; }
      }
    }
  }

  const long base = ((long)b * NN + n) * NB + off;
#pragma unroll 4
  for (int j = 0; j < K; ++j) {
    const float px = db[bestI[j] * 3 + 0];
    const float py = db[bestI[j] * 3 + 1];
    const float pz = db[bestI[j] * 3 + 2];
    const float rx = px - qx, ry = py - qy, rz = pz - qz;
    const float sq = rx * rx + ry * ry + rz * rz;
    const float dist = (sq > 0.0f) ? sqrtf(sq) : 0.0f;
    float4* fo = (float4*)&feats[(base + j) * 4];
    *fo = make_float4(rx, ry, rz, dist);
    grouped[(base + j) * 3 + 0] = px;
    grouped[(base + j) * 3 + 1] = py;
    grouped[(base + j) * 3 + 2] = pz;
  }
}

// ---------------- WMMA fragment loaders (ISA 7.12.2 layouts) ----------------
// strideU = row stride in 32-bit units (CO/2 = 32 for 64-wide rows, KP/2 = 16 for padded rows)
// A 16x32 f16: lane L -> M = m0 + (L&15); half = L>>4;
//   reg j<4 : K = kc + half*8 + 2j ; reg j>=4 : K = kc + 16 + half*8 + 2(j-4)
__device__ __forceinline__ v16h load_fragA(const _Float16* base, int m0, int kc, int lane,
                                           int strideU) {
  const unsigned* b32 = (const unsigned*)base;
  const int M = m0 + (lane & 15);
  const int half = lane >> 4;
  Frag16 f;
#pragma unroll
  for (int j = 0; j < 8; ++j) {
    const int kb = kc + ((j < 4) ? (half * 8 + j * 2) : (16 + half * 8 + (j - 4) * 2));
    f.u[j] = b32[M * strideU + (kb >> 1)];
  }
  return f.h;
}

// B 32x16 f16 with B(K,N) = base[N*stride + K] (row index = N, contiguous in K):
//   lane L -> N = n0 + (L&15); half = L>>4; same K order as A:
//   reg j<4 : K = kc + half*8 + 2j ; reg j>=4 : K = kc + 16 + half*8 + 2(j-4)
__device__ __forceinline__ v16h load_fragB(const _Float16* base, int n0, int kc, int lane,
                                           int strideU) {
  const unsigned* b32 = (const unsigned*)base;
  const int N = n0 + (lane & 15);
  const int half = lane >> 4;
  Frag16 f;
#pragma unroll
  for (int j = 0; j < 8; ++j) {
    const int kb = kc + ((j < 4) ? (half * 8 + j * 2) : (16 + half * 8 + (j - 4) * 2));
    f.u[j] = b32[N * strideU + (kb >> 1)];
  }
  return f.h;
}

__device__ __forceinline__ float redmax16(float v) {
#pragma unroll
  for (int m = 1; m < 16; m <<= 1) v = fmaxf(v, __shfl_xor(v, m));
  return v;
}
__device__ __forceinline__ float redsum16(float v) {
#pragma unroll
  for (int m = 1; m < 16; m <<= 1) v += __shfl_xor(v, m);
  return v;
}
__device__ __forceinline__ float redmax32(float v) {
#pragma unroll
  for (int m = 1; m < 32; m <<= 1) v = fmaxf(v, __shfl_xor(v, m));
  return v;
}
__device__ __forceinline__ float redsum32(float v) {
#pragma unroll
  for (int m = 1; m < 32; m <<= 1) v += __shfl_xor(v, m);
  return v;
}

// Projection GEMM on WMMA: dst(64x64) = Fpad(64x32) @ Wt^T, bias folded via ones-column.
// transposeStore=true writes dst[N*64+M] (for V stored channel-major).
__device__ __forceinline__ void proj_wmma(const _Float16* sF, const _Float16* sWt,
                                          _Float16* dst, bool transposeStore, int lane) {
  const int half = lane >> 4, col = lane & 15;
#pragma unroll
  for (int mt = 0; mt < 4; ++mt) {
    const v16h a = load_fragA(sF, mt * 16, 0, lane, KP / 2);
#pragma unroll
    for (int nt = 0; nt < 4; ++nt) {
      v8f c = {};
      c = __builtin_amdgcn_wmma_f32_16x16x32_f16(false, a, false,
            load_fragB(sWt, nt * 16, 0, lane, KP / 2), (short)0, c, false, false);
#pragma unroll
      for (int r = 0; r < 8; ++r) {
        const int M = mt * 16 + r + 8 * half;   // neighbor row
        const int N = nt * 16 + col;            // output channel
        if (transposeStore) dst[N * 64 + M] = (_Float16)c[r];
        else                dst[M * 64 + N] = (_Float16)c[r];
      }
    }
  }
}

// ---------------- attention: one wave32 per point, all GEMMs on WMMA ----------------
__global__ __launch_bounds__(32)
void attn_kernel(const float* __restrict__ feats, const float* __restrict__ grouped,
                 const float* __restrict__ Wq, const float* __restrict__ Wk,
                 const float* __restrict__ Wv,
                 const float* __restrict__ bq, const float* __restrict__ bk,
                 const float* __restrict__ bv,
                 float* __restrict__ out) {
  // sQ doubles as the attention-probability matrix after GEMM1 (each m-tile's
  // rows are consumed into A-fragments before being overwritten).
  __shared__ _Float16 sQ[NB * CO];    // Q rows, later attn rows ([nbr][co])
  __shared__ _Float16 sKm[NB * CO];   // K rows ([nbr][co])
  __shared__ _Float16 sVt[CO * NB];   // V transposed ([co][nbr])
  __shared__ _Float16 sF[NB * KP];    // padded feats: cols 0..3 feats, col 4 = 1.0, rest 0
  __shared__ _Float16 sWtQ[CO * KP];  // Wt[c][i] = W[i][c], col 4 = bias[c], rest 0
  __shared__ _Float16 sWtK[CO * KP];
  __shared__ _Float16 sWtV[CO * KP];
  __shared__ float sScore[NB];

  const int lane = threadIdx.x;
  const int n = blockIdx.x, b = blockIdx.y;
  const long pt = (long)b * NN + n;

  // ---- stage padded operands: zero-fill, then write the live columns ----
  {
    unsigned* zF = (unsigned*)sF;
    unsigned* zQ = (unsigned*)sWtQ;
    unsigned* zK = (unsigned*)sWtK;
    unsigned* zV = (unsigned*)sWtV;
#pragma unroll
    for (int i = lane; i < NB * KP / 2; i += 32) {
      zF[i] = 0u; zQ[i] = 0u; zK[i] = 0u; zV[i] = 0u;
    }
  }
  __syncthreads();
#pragma unroll
  for (int c = lane; c < CO; c += 32) {
#pragma unroll
    for (int i = 0; i < 4; ++i) {
      sWtQ[c * KP + i] = (_Float16)Wq[i * CO + c];
      sWtK[c * KP + i] = (_Float16)Wk[i * CO + c];
      sWtV[c * KP + i] = (_Float16)Wv[i * CO + c];
    }
    sWtQ[c * KP + 4] = (_Float16)bq[c];
    sWtK[c * KP + 4] = (_Float16)bk[c];
    sWtV[c * KP + 4] = (_Float16)bv[c];
  }
  {
    const float4* fr = (const float4*)(feats + pt * NB * 4);
#pragma unroll
    for (int rr = 0; rr < 2; ++rr) {
      const int row = lane + rr * 32;
      const float4 f = fr[row];
      sF[row * KP + 0] = (_Float16)f.x;
      sF[row * KP + 1] = (_Float16)f.y;
      sF[row * KP + 2] = (_Float16)f.z;
      sF[row * KP + 3] = (_Float16)f.w;
      sF[row * KP + 4] = (_Float16)1.0f;   // bias column
    }
  }
  __syncthreads();

  // ---- QKV projection on WMMA (48 wmma ops) ----
  proj_wmma(sF, sWtQ, sQ,  false, lane);
  proj_wmma(sF, sWtK, sKm, false, lane);
  proj_wmma(sF, sWtV, sVt, true,  lane);
  __syncthreads();

  const float scale = 0.125f; // 1/sqrt(64)
  const int half = lane >> 4, col = lane & 15;

  // ---- GEMM1: S = Q * K^T (64x64x64), fused row-softmax, attn -> sQ ----
#pragma unroll
  for (int mt = 0; mt < 4; ++mt) {
    const v16h a0 = load_fragA(sQ, mt * 16, 0, lane, CO / 2);
    const v16h a1 = load_fragA(sQ, mt * 16, 32, lane, CO / 2);
    v8f acc[4];
#pragma unroll
    for (int nt = 0; nt < 4; ++nt) {
      v8f c = {};
      c = __builtin_amdgcn_wmma_f32_16x16x32_f16(false, a0, false,
            load_fragB(sKm, nt * 16, 0, lane, CO / 2), (short)0, c, false, false);
      c = __builtin_amdgcn_wmma_f32_16x16x32_f16(false, a1, false,
            load_fragB(sKm, nt * 16, 32, lane, CO / 2), (short)0, c, false, false);
      acc[nt] = c;
    }
    // row softmax (row M = mt*16 + r + 8*half spans 4 nt frags x 16 lanes)
#pragma unroll
    for (int r = 0; r < 8; ++r) {
      const float x0 = acc[0][r] * scale, x1 = acc[1][r] * scale;
      const float x2 = acc[2][r] * scale, x3 = acc[3][r] * scale;
      float m = fmaxf(fmaxf(x0, x1), fmaxf(x2, x3));
      m = redmax16(m);
      const float e0 = __expf(x0 - m), e1 = __expf(x1 - m);
      const float e2 = __expf(x2 - m), e3 = __expf(x3 - m);
      const float inv = 1.0f / redsum16(e0 + e1 + e2 + e3);
      const int row = mt * 16 + r + 8 * half;
      sQ[row * CO +  0 + col] = (_Float16)(e0 * inv);
      sQ[row * CO + 16 + col] = (_Float16)(e1 * inv);
      sQ[row * CO + 32 + col] = (_Float16)(e2 * inv);
      sQ[row * CO + 48 + col] = (_Float16)(e3 * inv);
    }
  }
  __syncthreads();

  // ---- GEMM2: Y = A * V (64x64x64); score[row] = max over channels ----
#pragma unroll
  for (int mt = 0; mt < 4; ++mt) {
    const v16h a0 = load_fragA(sQ, mt * 16, 0, lane, CO / 2);
    const v16h a1 = load_fragA(sQ, mt * 16, 32, lane, CO / 2);
    v8f acc[4];
#pragma unroll
    for (int nt = 0; nt < 4; ++nt) {
      v8f c = {};
      c = __builtin_amdgcn_wmma_f32_16x16x32_f16(false, a0, false,
            load_fragB(sVt, nt * 16, 0, lane, CO / 2), (short)0, c, false, false);
      c = __builtin_amdgcn_wmma_f32_16x16x32_f16(false, a1, false,
            load_fragB(sVt, nt * 16, 32, lane, CO / 2), (short)0, c, false, false);
      acc[nt] = c;
    }
#pragma unroll
    for (int r = 0; r < 8; ++r) {
      float m = fmaxf(fmaxf(acc[0][r], acc[1][r]), fmaxf(acc[2][r], acc[3][r]));
      m = redmax16(m);
      if (col == 0) sScore[mt * 16 + r + 8 * half] = m;
    }
  }
  __syncthreads();

  // ---- softmax over 64 neighbor scores + weighted sum of grouped coords ----
  const float s0 = sScore[lane], s1 = sScore[lane + 32];
  const float m = redmax32(fmaxf(s0, s1));
  const float e0 = __expf(s0 - m), e1 = __expf(s1 - m);
  const float ssum = redsum32(e0 + e1);
  const float* g = grouped + pt * NB * 3;
  float gx = e0 * g[lane * 3 + 0] + e1 * g[(lane + 32) * 3 + 0];
  float gy = e0 * g[lane * 3 + 1] + e1 * g[(lane + 32) * 3 + 1];
  float gz = e0 * g[lane * 3 + 2] + e1 * g[(lane + 32) * 3 + 2];
  gx = redsum32(gx); gy = redsum32(gy); gz = redsum32(gz);
  if (lane == 0) {
    const float inv = 1.0f / ssum;
    out[((long)b * 3 + 0) * NN + n] = gx * inv;
    out[((long)b * 3 + 1) * NN + n] = gy * inv;
    out[((long)b * 3 + 2) * NN + n] = gz * inv;
  }
}

// ---------------- launch ----------------
extern "C" void kernel_launch(void* const* d_in, const int* in_sizes, int n_in,
                              void* d_out, int out_size, void* d_ws, size_t ws_size,
                              hipStream_t stream) {
  const float* p1 = (const float*)d_in[0];
  const float* p2 = (const float*)d_in[1];
  const float* p3 = (const float*)d_in[2];
  const float* Wq = (const float*)d_in[3];
  const float* Wk = (const float*)d_in[4];
  const float* Wv = (const float*)d_in[5];
  const float* bq = (const float*)d_in[6];
  const float* bk = (const float*)d_in[7];
  const float* bv = (const float*)d_in[8];
  const int* ridx1 = (const int*)d_in[9];
  const int* ridx2 = (const int*)d_in[10];
  // d_in[11] = k (32), d_in[12] = t (0.5): fixed by harness, baked in.
  float* out = (float*)d_out;

  const size_t PN = (size_t)BB * NN * 3;           // 98304 floats
  char* ws = (char*)d_ws;
  float* p1t  = (float*)ws;                 ws += PN * sizeof(float);
  float* p2t  = (float*)ws;                 ws += PN * sizeof(float);
  float* p3t  = (float*)ws;                 ws += PN * sizeof(float);
  float* newp = (float*)ws;                 ws += PN * sizeof(float);
  float* feats   = (float*)ws;              ws += (size_t)BB * NN * NB * 4 * sizeof(float);
  float* grouped = (float*)ws;              /* ws += BB*NN*NB*3*4 */

  prep_kernel<<<dim3(NN / 256, BB), 256, 0, stream>>>(p1, p2, p3, ridx1, ridx2,
                                                      p1t, p2t, p3t, newp);

  knn_kernel<16><<<dim3(NN / 128, BB), 128, 0, stream>>>(newp, p1t, feats, grouped, 0);
  knn_kernel<16><<<dim3(NN / 128, BB), 128, 0, stream>>>(newp, p2t, feats, grouped, 16);
  knn_kernel<32><<<dim3(NN / 128, BB), 128, 0, stream>>>(newp, p3t, feats, grouped, 32);

  attn_kernel<<<dim3(NN, BB), 32, 0, stream>>>(feats, grouped, Wq, Wk, Wv, bq, bk, bv, out);
}
